// Disentangled_MHSA_87565793231071
// MI455X (gfx1250) — matrix operand order repaired and checked by reading, not compiled, not run on hardware
//
#include <hip/hip_runtime.h>
#include <hip/hip_bf16.h>

typedef __attribute__((ext_vector_type(16))) __bf16        v16bf;
typedef __attribute__((ext_vector_type(8)))  float         v8f;
typedef __attribute__((ext_vector_type(4)))  float         f32x4;
typedef __attribute__((ext_vector_type(4)))  unsigned int  u32x4;
typedef __attribute__((ext_vector_type(8)))  int           i32x8;
typedef __attribute__((ext_vector_type(4)))  int           i32x4;

#define BB   2
#define SS   2048
#define DD   128
#define HD   1024      // H*D
#define HH   8
#define MAXPOS 512
#define NEGV (-1.0e9f)
#define SCALE 0.08838834764831845f   // 1/sqrt(128)

#if __has_builtin(__builtin_amdgcn_tensor_load_to_lds) && \
    __has_builtin(__builtin_amdgcn_s_wait_tensorcnt)
#define HAVE_TDM 1
#else
#define HAVE_TDM 0
#endif

union BF16Frag { v16bf v; __bf16 h[16]; unsigned short u[16]; };

__device__ __forceinline__ unsigned short f2bf(float f) {
  __bf16 h = (__bf16)f;                        // native v_cvt to bf16
  return __builtin_bit_cast(unsigned short, h);
}

// ---------------------------------------------------------------------------
// Kernel 1: fused QKV projection  X[4096,128] @ W[128,1024] -> bf16 [B,H,S,D]
// One wave per 16x16 output tile; K=128 as 4 x wmma_f32_16x16x32_bf16.
// ---------------------------------------------------------------------------
__global__ __launch_bounds__(128)
void qkv_proj_wmma(const float* __restrict__ q, const float* __restrict__ k,
                   const float* __restrict__ v, const float* __restrict__ Wq,
                   const float* __restrict__ Wk, const float* __restrict__ Wv,
                   unsigned short* __restrict__ Qb, unsigned short* __restrict__ Kb,
                   unsigned short* __restrict__ Vb) {
  const int lane = threadIdx.x & 31;
  const int wave = threadIdx.x >> 5;
  const int z = blockIdx.z;
  const float* X = (z == 0) ? q : (z == 1) ? k : v;
  const float* W = (z == 0) ? Wq : (z == 1) ? Wk : Wv;
  unsigned short* O = (z == 0) ? Qb : (z == 1) ? Kb : Vb;

  const int tile = blockIdx.x * 4 + wave;       // 16384 tiles total
  const int m0 = (tile >> 6) * 16;              // 256 M-tiles
  const int n0 = (tile & 63) * 16;              // 64  N-tiles

  const int halfsel = (lane < 16) ? 0 : 1;
  const int lrow = lane & 15;
  const int koff = halfsel * 8;                 // A-fragment K split
  const int klo  = halfsel * 16;                // unused; B fragment also uses koff

  v8f acc = {};
#pragma unroll
  for (int kc = 0; kc < 4; ++kc) {
    BF16Frag a, b;
    const float* arow = X + (m0 + lrow) * DD + kc * 32;
    f32x4 p0 = *(const f32x4*)(arow + koff);
    f32x4 p1 = *(const f32x4*)(arow + koff + 4);
    f32x4 p2 = *(const f32x4*)(arow + 16 + koff);
    f32x4 p3 = *(const f32x4*)(arow + 16 + koff + 4);
#pragma unroll
    for (int j = 0; j < 4; ++j) {
      a.h[j]      = (__bf16)p0[j];
      a.h[4 + j]  = (__bf16)p1[j];
      a.h[8 + j]  = (__bf16)p2[j];
      a.h[12 + j] = (__bf16)p3[j];
    }
    const float* wcol = W + n0 + lrow;
#pragma unroll
    for (int j = 0; j < 16; ++j)
      b.h[j] = (__bf16)wcol[(kc * 32 + (j < 8 ? koff + j : 8 + koff + j)) * HD];
    acc = __builtin_amdgcn_wmma_f32_16x16x32_bf16(false, a.v, false, b.v,
                                                  (short)0, acc, false, false);
  }

  // scatter-store into [B,H,S,D] bf16
  const int n  = n0 + lrow;
  const int h  = n >> 7;
  const int dd = n & 127;
#pragma unroll
  for (int r = 0; r < 8; ++r) {
    const int row  = m0 + halfsel * 8 + r;      // global M (= b*S + s)
    const int bidx = row >> 11;
    const int s    = row & 2047;
    O[(((size_t)(bidx * HH + h) * SS + s) * DD) + dd] = f2bf(acc[r]);
  }
}

// ---------------------------------------------------------------------------
// Kernel 2: causal + RPE flash attention, bf16 WMMA, fp32 online softmax.
// 4 waves/WG; wave owns 16 queries; 32-key tiles staged in LDS.
// K tile staged via the Tensor Data Mover (tensor_load_to_lds) when available.
// ---------------------------------------------------------------------------
__global__ __launch_bounds__(128)
void flash_attn_wmma(const unsigned short* __restrict__ Qb,
                     const unsigned short* __restrict__ Kb,
                     const unsigned short* __restrict__ Vb,
                     const float* __restrict__ emb,
                     float* __restrict__ out) {
  __shared__ unsigned short Ks[32 * 128];       // K tile   [key][d]
  __shared__ unsigned short Vts[128 * 32];      // V tile^T [d][key]
  __shared__ unsigned short Ps[4][16 * 32];     // per-wave P tile [m][k]

  const int lane = threadIdx.x & 31;
  const int wave = threadIdx.x >> 5;
  const int bh = blockIdx.y;                    // b*H + h
  const int q0 = blockIdx.x * 64 + wave * 16;

  const unsigned short* Qh = Qb + (size_t)bh * SS * DD;
  const unsigned short* Kh = Kb + (size_t)bh * SS * DD;
  const unsigned short* Vh = Vb + (size_t)bh * SS * DD;

  const int halfsel = (lane < 16) ? 0 : 1;
  const int lrow = lane & 15;
  const int koff = halfsel * 8;
  const int klo  = halfsel * 16;

  // Q fragments for the 4 K-chunks (held in registers for the whole pass)
  BF16Frag aQ[4];
#pragma unroll
  for (int c = 0; c < 4; ++c) {
    const unsigned short* qrow = Qh + (size_t)(q0 + lrow) * DD + c * 32;
    ((u32x4*)&aQ[c])[0] = *(const u32x4*)(qrow + koff);
    ((u32x4*)&aQ[c])[1] = *(const u32x4*)(qrow + 16 + koff);
  }

  float m_run[8], l_run[8];
  v8f acc[8] = {};
#pragma unroll
  for (int r = 0; r < 8; ++r) { m_run[r] = -3.0e38f; l_run[r] = 0.f; }

  const int nkt = (blockIdx.x + 1) * 2;         // causal bound for this WG
  for (int kt = 0; kt < nkt; ++kt) {
    const int kbase = kt * 32;

#if HAVE_TDM
    // ---- TDM: DMA the 32x128 bf16 K tile into LDS (one issue per WG)
    if (wave == 0) {
      const unsigned lds_off = (unsigned)(uintptr_t)(&Ks[0]);
      const unsigned long long ga =
          (unsigned long long)(uintptr_t)(Kh + (size_t)kbase * DD);
      u32x4 g0;
      g0[0] = 1u;                                     // count=1 (user D#)
      g0[1] = lds_off;                                // lds_addr
      g0[2] = (unsigned)(ga & 0xffffffffu);           // global_addr[31:0]
      g0[3] = ((unsigned)(ga >> 32) & 0x1ffffffu)     // global_addr[56:32]
              | (2u << 30);                           // type=2 (image)
      i32x8 g1;
      g1[0] = (1 << 16);          // workgroup_mask=0, data_size=1 (2 bytes)
      g1[1] = (DD << 16);         // tensor_dim0=128 (bits 79:48, low half)
      g1[2] = (SS << 16);         // tensor_dim1=2048 (bits 111:80, low half)
      g1[3] = (DD << 16);         // tile_dim0=128 (bits 127:112)
      g1[4] = 32;                 // tile_dim1=32, tile_dim2=0
      g1[5] = DD;                 // tensor_dim0_stride=128 (low 32)
      g1[6] = 0;
      g1[7] = 0;
      i32x4 z4 = {0, 0, 0, 0};
      i32x8 z8 = {0, 0, 0, 0, 0, 0, 0, 0};
      __builtin_amdgcn_tensor_load_to_lds(g0, g1, z4, z4, z8, 0);
    }
#else
    // ---- fallback: cooperative K tile copy
    {
      const int t = threadIdx.x;
      const int key = t >> 2;
      const int dpart = (t & 3) * 32;
      const u32x4* gk = (const u32x4*)(Kh + (size_t)(kbase + key) * DD + dpart);
      u32x4* sk = (u32x4*)(Ks + key * DD + dpart);
#pragma unroll
      for (int i = 0; i < 4; ++i) sk[i] = gk[i];
    }
#endif

    // ---- cooperative transposed V tile staging (all 128 threads)
    {
      const int t = threadIdx.x;
      const int key = t >> 2;                   // 0..31
      const int dpart = (t & 3) * 32;
      const unsigned short* gv = Vh + (size_t)(kbase + key) * DD + dpart;
#pragma unroll
      for (int j = 0; j < 32; ++j) Vts[(dpart + j) * 32 + key] = gv[j];
    }
    if (kt + 1 < nkt) {                         // global_prefetch_b8 next tiles
      __builtin_prefetch(Kh + (size_t)(kbase + 32) * DD, 0, 1);
      __builtin_prefetch(Vh + (size_t)(kbase + 32) * DD, 0, 1);
    }
#if HAVE_TDM
    if (wave == 0) __builtin_amdgcn_s_wait_tensorcnt(0);
#endif
    __syncthreads();

    // ---- scores S = Q K^T for two 16-key halves (8 WMMAs)
    float sv[2][8];
#pragma unroll
    for (int hh = 0; hh < 2; ++hh) {
      v8f s = {};
      const int keyl = hh * 16 + lrow;
#pragma unroll
      for (int c = 0; c < 4; ++c) {
        BF16Frag b;
        const u32x4* kp = (const u32x4*)(Ks + keyl * DD + c * 32 + koff);
        ((u32x4*)&b)[0] = kp[0];
        ((u32x4*)&b)[1] = kp[2];
        s = __builtin_amdgcn_wmma_f32_16x16x32_bf16(false, aQ[c].v, false, b.v,
                                                    (short)0, s, false, false);
      }
      // causal mask + relative-position bias (diagonal is bias-masked too)
      const int n = kbase + keyl;
#pragma unroll
      for (int r = 0; r < 8; ++r) {
        const int m = q0 + halfsel * 8 + r;
        float x = s[r] * SCALE;
        if (n > m) x = NEGV;
        const int dist = n - m;
        float bias = NEGV;
        if (dist < 0 && dist > -MAXPOS) bias = emb[-dist - 1];
        sv[hh][r] = x + bias;
      }
    }

    // ---- online softmax (row reductions across 16-lane halves)
    float alpha[8], p0[8], p1[8];
#pragma unroll
    for (int r = 0; r < 8; ++r) {
      float mx = fmaxf(sv[0][r], sv[1][r]);
      mx = fmaxf(mx, __shfl_xor(mx, 1));
      mx = fmaxf(mx, __shfl_xor(mx, 2));
      mx = fmaxf(mx, __shfl_xor(mx, 4));
      mx = fmaxf(mx, __shfl_xor(mx, 8));
      const float mnew = fmaxf(m_run[r], mx);
      alpha[r] = __expf(m_run[r] - mnew);
      p0[r] = __expf(sv[0][r] - mnew);
      p1[r] = __expf(sv[1][r] - mnew);
      float rs = p0[r] + p1[r];
      rs += __shfl_xor(rs, 1);
      rs += __shfl_xor(rs, 2);
      rs += __shfl_xor(rs, 4);
      rs += __shfl_xor(rs, 8);
      l_run[r] = l_run[r] * alpha[r] + rs;
      m_run[r] = mnew;
    }
#pragma unroll
    for (int ch = 0; ch < 8; ++ch)
#pragma unroll
      for (int r = 0; r < 8; ++r) acc[ch][r] *= alpha[r];

    // ---- re-layout P (C-layout f32 -> A-layout bf16) via per-wave LDS
    unsigned short* pw = Ps[wave];
#pragma unroll
    for (int r = 0; r < 8; ++r) {
      const int m = halfsel * 8 + r;
      pw[m * 32 + lrow]      = f2bf(p0[r]);
      pw[m * 32 + 16 + lrow] = f2bf(p1[r]);
    }
    BF16Frag aP;
    {
      const unsigned short* prow = pw + lrow * 32;
      ((u32x4*)&aP)[0] = *(const u32x4*)(prow + koff);
      ((u32x4*)&aP)[1] = *(const u32x4*)(prow + 16 + koff);
    }

    // ---- O += P V  (8 WMMAs over the 128-wide d dimension)
#pragma unroll
    for (int ch = 0; ch < 8; ++ch) {
      BF16Frag bv;
      const u32x4* vp = (const u32x4*)(Vts + (ch * 16 + lrow) * 32 + koff);
      ((u32x4*)&bv)[0] = vp[0];
      ((u32x4*)&bv)[1] = vp[2];
      acc[ch] = __builtin_amdgcn_wmma_f32_16x16x32_bf16(false, aP.v, false, bv.v,
                                                        (short)0, acc[ch], false, false);
    }
    __syncthreads();
  }

  // ---- normalize and store out[b, s, h*D + d]
  const int bidx = bh >> 3;
  const int h = bh & 7;
#pragma unroll
  for (int r = 0; r < 8; ++r) {
    const int m = q0 + halfsel * 8 + r;
    const float invl = 1.0f / l_run[r];
#pragma unroll
    for (int ch = 0; ch < 8; ++ch) {
      const int dd = ch * 16 + lrow;
      out[((size_t)bidx * SS + m) * HD + h * DD + dd] = acc[ch][r] * invl;
    }
  }
}

// ---------------------------------------------------------------------------
extern "C" void kernel_launch(void* const* d_in, const int* in_sizes, int n_in,
                              void* d_out, int out_size, void* d_ws, size_t ws_size,
                              hipStream_t stream) {
  const float* q   = (const float*)d_in[0];
  const float* k   = (const float*)d_in[1];
  const float* v   = (const float*)d_in[2];
  const float* Wq  = (const float*)d_in[3];
  const float* Wk  = (const float*)d_in[4];
  const float* Wv  = (const float*)d_in[5];
  const float* emb = (const float*)d_in[6];
  float* out = (float*)d_out;

  const size_t n_bf = (size_t)BB * HH * SS * DD;   // 4,194,304 elems
  unsigned short* Qb = (unsigned short*)d_ws;
  unsigned short* Kb = Qb + n_bf;
  unsigned short* Vb = Kb + n_bf;

  // 16384 tiles per matrix / 4 waves per block = 4096 blocks; z selects q/k/v
  qkv_proj_wmma<<<dim3(4096, 1, 3), dim3(128), 0, stream>>>(
      q, k, v, Wq, Wk, Wv, Qb, Kb, Vb);

  // S/64 query blocks x (B*H) heads
  flash_attn_wmma<<<dim3(SS / 64, BB * HH), dim3(128), 0, stream>>>(
      Qb, Kb, Vb, emb, out);
}
